// HMLSTM_6657199309458
// MI455X (gfx1250) — matrix operand order repaired and checked by reading, not compiled, not run on hardware
//
#include <hip/hip_runtime.h>
#include <hip/hip_bf16.h>
#include <stdint.h>

// ---------------- types ----------------
typedef __bf16 bf16_t;
typedef bf16_t v16bf __attribute__((ext_vector_type(16)));
typedef float  v16f  __attribute__((ext_vector_type(16)));
typedef float  v8f   __attribute__((ext_vector_type(8)));
typedef float  f4    __attribute__((ext_vector_type(4)));

#define B_   64
#define T_   512
#define H_   512
#define G_   2049
#define HSEQ_ ((size_t)B_ * T_ * 1536)

// ---------------- helpers ----------------
__device__ __forceinline__ float sigm_(float x) { return 1.f / (1.f + __expf(-x)); }
__device__ __forceinline__ float tanh_(float x) {
  float e = __expf(-2.f * x);
  return (1.f - e) / (1.f + e);
}

// CDNA5 async global->LDS copy (ASYNCcnt-tracked)
__device__ __forceinline__ void async_ld_b128(uint32_t ldsoff, unsigned long long ga) {
  asm volatile("global_load_async_to_lds_b128 %0, %1, off"
               :: "v"(ldsoff), "v"(ga) : "memory");
}
__device__ __forceinline__ void wait_async_le2() { asm volatile("s_wait_asynccnt 2" ::: "memory"); }
__device__ __forceinline__ void wait_async_le0() { asm volatile("s_wait_asynccnt 0" ::: "memory"); }

// ---------------- weight pack: [Wb;Wr;Wt] f32 -> bf16 B-fragment layout ----------------
__global__ void pack_weights_kernel(const float* __restrict__ Wb,
                                    const float* __restrict__ Wr,
                                    const float* __restrict__ Wt,
                                    bf16_t* __restrict__ dst, int Kt) {
  int total = 128 * Kt * 512;
  int tid = blockIdx.x * 256 + threadIdx.x;
  if (tid >= total) return;
  int e    = tid & 511;
  int tile = tid >> 9;
  int kt   = tile % Kt;
  int nt   = tile / Kt;
  int lane = e >> 4, v = e & 15;
  int n = nt * 16 + (lane & 15);
  int k = kt * 32 + (lane >> 4) * 16 + v;
  int seg = k >> 9, kk = k & 511;
  const float* W = (seg == 0) ? Wb : (seg == 1 ? Wr : Wt);
  dst[tid] = (bf16_t)W[(size_t)kk * G_ + n];
}

__global__ void zero_kernel(float* p, int n) {
  int i = blockIdx.x * 256 + threadIdx.x;
  if (i < n) p[i] = 0.f;
}

// ---------------- fused layer-step kernel ----------------
// grid = 9 x 256. blocks 0..7: 64-col gate slices (all 4 gates, fused LSTM epilogue),
// block 8: z_tilde column. KT = #K chunks of 32, SEGS = #A segments.
template <int KT, int SEGS>
__global__ void __launch_bounds__(256)
hmlstm_step_kernel(const float* __restrict__ hb, int hb_stride,
                   const float* __restrict__ hprev,
                   const float* __restrict__ htop,
                   const float* __restrict__ zb,     // null -> 1.0
                   const float* __restrict__ zl,
                   const bf16_t* __restrict__ Wp,
                   const float* __restrict__ bias,
                   const float* __restrict__ Wb_full,
                   const float* __restrict__ Wr_full,
                   const float* __restrict__ Wt_full,
                   float* __restrict__ c_st,
                   float* __restrict__ h_out,
                   float* __restrict__ z_out,
                   float* __restrict__ out,
                   int t, int l) {
  extern __shared__ float smem[];
  const int tid = threadIdx.x;
  const int blk = blockIdx.x;

  if (blk == 8) {
    // ---------- z_tilde column ----------
    float* zw = smem; // [SEGS][512]
    for (int idx = tid; idx < SEGS * H_; idx += 256) {
      int seg = idx >> 9, kk = idx & 511;
      const float* W = (seg == 0) ? Wb_full : (seg == 1 ? Wr_full : Wt_full);
      zw[idx] = W[(size_t)kk * G_ + 2048];
    }
    __syncthreads();
    int m = tid >> 2, part = tid & 3;
    const float* hbr = hb + (size_t)m * hb_stride;
    const float* hpr = hprev + m * H_;
    const float* htr = (SEGS == 3) ? (htop + m * H_) : nullptr;
    float s0 = 0.f, s1 = 0.f, s2 = 0.f;
    int k0 = part * 128;
#pragma unroll 2
    for (int q = 0; q < 128; ++q) {
      int k = k0 + q;
      s0 += hbr[k] * zw[k];
      s1 += hpr[k] * zw[H_ + k];
      if (SEGS == 3) s2 += htr[k] * zw[2 * H_ + k];
    }
    float zbm = zb ? zb[m] : 1.0f;
    float zlm = zl[m];
    float g = zbm * s0 + s1 + zlm * s2;
    g += __shfl_down(g, 2, 4);
    g += __shfl_down(g, 1, 4);
    if (part == 0) {
      g += bias[2048];
      float zh = 0.5f * (g + 1.0f);
      zh = fminf(fmaxf(zh, 0.f), 1.f);
      float z = __builtin_rintf(zh);
      z_out[m] = z;
      out[HSEQ_ + ((size_t)m * T_ + t) * 3 + l] = z;
    }
    return;
  }

  // ---------- gate GEMM path ----------
  const int lane  = tid & 31;
  const int w     = tid >> 5;
  const int mtile = w >> 1;
  const int jhalf = w & 1;
  const int half  = lane >> 4;

  const uint32_t ldsbase = __builtin_amdgcn_groupstaticsize();

  const int   arow   = mtile * 16 + (lane & 15);
  const float zb_row = zb ? zb[arow] : 1.0f;
  const float zl_row = zl[arow];

  // flattened 8 output tiles: idx = g*2 + jj
  v8f acc[8];
#pragma unroll
  for (int i = 0; i < 8; ++i)
    acc[i] = (v8f){0.f, 0.f, 0.f, 0.f, 0.f, 0.f, 0.f, 0.f};

  const v16bf* Wv = (const v16bf*)Wp;
  const v16bf* tp[8];
#pragma unroll
  for (int i = 0; i < 8; ++i) {
    int g = i >> 1, jj = i & 1;
    int ntile = g * 32 + blk * 4 + jhalf * 2 + jj;
    tp[i] = Wv + (size_t)ntile * KT * 32 + lane;
  }

  // per-segment A-staging base addresses, arrays indexed ONLY by unrolled constants
  // (selection by value-ternary -> v_cndmask, no scratch spill)
  unsigned long long sA0[2], sA1[2], sA2[2];
  uint32_t           ldsA[2];
#pragma unroll
  for (int j = 0; j < 2; ++j) {
    int ti  = tid * 2 + j;
    int row = ti >> 3;
    int q   = ti & 7;
    sA0[j] = (unsigned long long)(uintptr_t)(hb + (size_t)row * hb_stride + q * 4);
    sA1[j] = (unsigned long long)(uintptr_t)(hprev + row * H_ + q * 4);
    sA2[j] = (SEGS == 3) ? (unsigned long long)(uintptr_t)(htop + row * H_ + q * 4)
                         : sA1[j];
    ldsA[j] = ldsbase + (uint32_t)(row * 144 + q * 16);
  }

  auto stage = [&](int c, int buf) {
    int seg = c >> 4;
    uint32_t koff = (uint32_t)(c & 15) * 128u;
    uint32_t lb = (uint32_t)buf * 9216u;
#pragma unroll
    for (int j = 0; j < 2; ++j) {
      unsigned long long base = (seg == 0) ? sA0[j] : ((seg == 1) ? sA1[j] : sA2[j]);
      async_ld_b128(ldsA[j] + lb, base + koff);
    }
  };

  stage(0, 0);
#pragma unroll
  for (int seg = 0; seg < SEGS; ++seg) {
    const float sA = (seg == 0) ? zb_row : ((seg == 1) ? 1.0f : zl_row);
#pragma unroll 4
    for (int ck = 0; ck < 16; ++ck) {
      const int c = seg * 16 + ck;
      const size_t coff = (size_t)c * 32;  // v16bf units -> +c*1024B immediate

      // issue ALL B-fragment loads first; latency hidden behind async-wait + barrier + A cvt
      v16bf barr[8];
#pragma unroll
      for (int i = 0; i < 8; ++i) barr[i] = tp[i][coff];

      if (c + 1 < KT) { stage(c + 1, (c + 1) & 1); wait_async_le2(); }
      else            { wait_async_le0(); }
      __syncthreads();

      // A fragment from LDS f32 chunk, scale + hw cvt to bf16
      const float* As = smem + (c & 1) * 2304;
      const f4* ap = (const f4*)(As + (size_t)(mtile * 16 + (lane & 15)) * 36 + half * 16);
      f4 a0 = ap[0];   // K rel 0..3
      f4 a1 = ap[1];   // K rel 4..7
      f4 a2 = ap[2];   // K rel 8..11
      f4 a3 = ap[3];   // K rel 12..15
      v16f afv;
      afv[0]=a0[0];  afv[1]=a0[1];  afv[2]=a0[2];  afv[3]=a0[3];
      afv[4]=a1[0];  afv[5]=a1[1];  afv[6]=a1[2];  afv[7]=a1[3];
      afv[8]=a2[0];  afv[9]=a2[1];  afv[10]=a2[2]; afv[11]=a2[3];
      afv[12]=a3[0]; afv[13]=a3[1]; afv[14]=a3[2]; afv[15]=a3[3];
      v16bf av = __builtin_convertvector(afv * sA, v16bf);

#pragma unroll
      for (int i = 0; i < 8; ++i) {
        acc[i] = __builtin_amdgcn_wmma_f32_16x16x32_bf16(
            false, av, false, barr[i], (short)0, acc[i], false, false);
      }
      __syncthreads();
    }
  }

  // ---------- fused LSTM-cell epilogue ----------
  const int mrow0 = mtile * 16 + half * 8;
  f4 zl_lo = *(const f4*)(zl + mrow0);
  f4 zl_hi = *(const f4*)(zl + mrow0 + 4);
  f4 zb_lo = zb ? *(const f4*)(zb + mrow0)     : (f4){1.f, 1.f, 1.f, 1.f};
  f4 zb_hi = zb ? *(const f4*)(zb + mrow0 + 4) : (f4){1.f, 1.f, 1.f, 1.f};

#pragma unroll
  for (int jj = 0; jj < 2; ++jj) {
    int j  = blk * 64 + (jhalf * 2 + jj) * 16 + (lane & 15);
    float bf_ = bias[j];
    float bi_ = bias[512 + j];
    float bo_ = bias[1024 + j];
    float bg_ = bias[1536 + j];
#pragma unroll
    for (int e = 0; e < 8; ++e) {
      int m = mrow0 + e;
      float fg = sigm_(acc[0 * 2 + jj][e] + bf_);
      float ig = sigm_(acc[1 * 2 + jj][e] + bi_);
      float og = sigm_(acc[2 * 2 + jj][e] + bo_);
      float gg = tanh_(acc[3 * 2 + jj][e] + bg_);
      float zlm = (e < 4) ? zl_lo[e & 3] : zl_hi[e & 3];
      float zbm = (e < 4) ? zb_lo[e & 3] : zb_hi[e & 3];
      float copy = (1.f - zlm) * (1.f - zbm);
      float upd  = (1.f - zlm) * zbm;
      size_t idx = (size_t)m * H_ + j;
      float cp = c_st[idx];
      float hp = hprev[idx];
      float igg = ig * gg;
      float cn = zlm * igg + copy * cp + upd * (fg * cp + igg);
      float hn = copy * hp + (1.f - copy) * (og * tanh_(cn));
      c_st[idx]  = cn;
      h_out[idx] = hn;
      out[((size_t)m * T_ + t) * 1536 + (size_t)l * 512 + j] = hn;
    }
  }
}

// ---------------- host ----------------
extern "C" void kernel_launch(void* const* d_in, const int* in_sizes, int n_in,
                              void* d_out, int out_size, void* d_ws, size_t ws_size,
                              hipStream_t stream) {
  (void)in_sizes; (void)n_in; (void)out_size; (void)ws_size;
  const float* x = (const float*)d_in[0];
  const float* Wb[3] = {(const float*)d_in[1], (const float*)d_in[5], (const float*)d_in[9]};
  const float* Wr[3] = {(const float*)d_in[2], (const float*)d_in[6], (const float*)d_in[10]};
  const float* Wt[3] = {(const float*)d_in[3], (const float*)d_in[7], (const float*)d_in[11]};
  const float* bs[3] = {(const float*)d_in[4], (const float*)d_in[8], (const float*)d_in[12]};

  char* ws = (char*)d_ws;
  const int Ktl[3] = {48, 48, 32};
  bf16_t* Wp[3];
  size_t off = 0;
  Wp[0] = (bf16_t*)(ws + off); off += (size_t)128 * 48 * 512 * 2;
  Wp[1] = (bf16_t*)(ws + off); off += (size_t)128 * 48 * 512 * 2;
  Wp[2] = (bf16_t*)(ws + off); off += (size_t)128 * 32 * 512 * 2;
  float* hbase = (float*)(ws + off);                 // [2][3][64*512]
  float* cbase = hbase + 2 * 3 * (B_ * H_);          // [3][64*512]
  float* zbase = cbase + 3 * (B_ * H_);              // [2][3][64]

  for (int l = 0; l < 3; ++l) {
    int total = 128 * Ktl[l] * 512;
    pack_weights_kernel<<<(total + 255) / 256, 256, 0, stream>>>(Wb[l], Wr[l], Wt[l], Wp[l], Ktl[l]);
  }
  int zn = 2 * 3 * (B_ * H_) + 3 * (B_ * H_) + 2 * 3 * B_;
  zero_kernel<<<(zn + 255) / 256, 256, 0, stream>>>(hbase, zn);

  float* out = (float*)d_out;
  const size_t HL = (size_t)B_ * H_;
  const int SMEM = 2 * 9216;

  for (int t = 0; t < T_; ++t) {
    int p = t & 1;
    float* hcur = hbase + (size_t)p * 3 * HL;
    float* hnxt = hbase + (size_t)(1 - p) * 3 * HL;
    float* zcur = zbase + p * 3 * B_;
    float* znxt = zbase + (1 - p) * 3 * B_;
    for (int l = 0; l < 3; ++l) {
      const float* hb; int hbs;
      if (l == 0) { hb = x + (size_t)t * 512; hbs = T_ * 512; }
      else        { hb = hnxt + (size_t)(l - 1) * HL; hbs = H_; }
      const float* hpv = hcur + (size_t)l * HL;
      const float* htv = (l < 2) ? (hcur + (size_t)(l + 1) * HL) : nullptr;
      const float* zbv = (l == 0) ? nullptr : (znxt + (l - 1) * B_);
      const float* zlv = zcur + l * B_;
      float* cst = cbase + (size_t)l * HL;
      float* hov = hnxt + (size_t)l * HL;
      float* zov = znxt + l * B_;
      if (l < 2) {
        hmlstm_step_kernel<48, 3><<<9, 256, SMEM, stream>>>(
            hb, hbs, hpv, htv, zbv, zlv, Wp[l], bs[l], Wb[l], Wr[l], Wt[l],
            cst, hov, zov, out, t, l);
      } else {
        hmlstm_step_kernel<32, 2><<<9, 256, SMEM, stream>>>(
            hb, hbs, hpv, htv, zbv, zlv, Wp[l], bs[l], Wb[l], Wr[l], Wt[l],
            cst, hov, zov, out, t, l);
      }
    }
  }
}
